// Attention_43963285242540
// MI455X (gfx1250) — matrix operand order repaired and checked by reading, not compiled, not run on hardware
//
#include <hip/hip_runtime.h>
#include <hip/hip_bf16.h>
#include <math.h>

#define B_   2
#define L_   2048
#define D_   2048
#define H_   32
#define KV_  8
#define G_   4        // H_/KV_
#define HD_  64
#define NQ_  (H_*HD_)   // 2048
#define NK_  (KV_*HD_)  // 512
#define M_   (B_*L_)    // 4096

typedef unsigned short u16;
typedef __attribute__((ext_vector_type(16))) __bf16 bf16x16;
typedef __attribute__((ext_vector_type(8)))  float  f32x8;

union BFrag { uint4 u[2]; bf16x16 v; };

__device__ __forceinline__ u16 f2bf(float f) {
  union { float f; unsigned u; } x; x.f = f;
  unsigned r = x.u + 0x7FFFu + ((x.u >> 16) & 1u);   // round-to-nearest-even
  return (u16)(r >> 16);
}

// ---------------- elementwise casts ----------------
__global__ void cast_f32_bf16(const float* __restrict__ in, u16* __restrict__ out, int n) {
  for (int i = blockIdx.x * blockDim.x + threadIdx.x; i < n; i += gridDim.x * blockDim.x)
    out[i] = f2bf(in[i]);
}

// V: [b][l][kv][hd] f32  ->  [b][kv][hd][l] bf16 (dim-major so PV B-frags are contiguous)
__global__ void v_post(const float* __restrict__ vf, u16* __restrict__ vT, int n) {
  for (int i = blockIdx.x * blockDim.x + threadIdx.x; i < n; i += gridDim.x * blockDim.x) {
    int hd  = i % HD_;
    int kvh = (i / HD_) % KV_;
    int l   = (i / (HD_ * KV_)) % L_;
    int b   =  i / (HD_ * KV_ * L_);
    vT[(((size_t)(b * KV_ + kvh)) * HD_ + hd) * L_ + l] = f2bf(vf[i]);
  }
}

// ---------------- bf16 NT GEMM: C(MxN,f32) = A(MxK,bf16) * W(NxK,bf16)^T ----------------
// 128 threads = 4 waves; each wave owns one 32(M) x 64(N) tile (2 A-frags x 4 B-frags,
// 8 WMMAs per 32-wide k-step; B-frags reused across both M sub-tiles in registers).
__global__ __launch_bounds__(128) void gemm_bf16_nt(
    const u16* __restrict__ A, const u16* __restrict__ W, float* __restrict__ C,
    int M, int N, int K) {
  const int lane = threadIdx.x & 31;
  const int wv   = threadIdx.x >> 5;
  const int lo   = lane & 15, hi = lane >> 4;
  const int m0 = blockIdx.y * 128 + wv * 32;
  const int n0 = blockIdx.x * 64;

  const u16* arow0 = A + (size_t)(m0 + lo)      * K + 16 * hi;
  const u16* arow1 = A + (size_t)(m0 + 16 + lo) * K + 16 * hi;
  const u16* brow0 = W + (size_t)(n0 +  0 + lo) * K + 16 * hi;
  const u16* brow1 = W + (size_t)(n0 + 16 + lo) * K + 16 * hi;
  const u16* brow2 = W + (size_t)(n0 + 32 + lo) * K + 16 * hi;
  const u16* brow3 = W + (size_t)(n0 + 48 + lo) * K + 16 * hi;

  f32x8 acc[2][4] = {};
  for (int k0 = 0; k0 < K; k0 += 32) {
    // speculative prefetch two k-steps ahead (global_prefetch_b8)
    __builtin_prefetch((const void*)(arow0 + k0 + 64), 0, 1);
    __builtin_prefetch((const void*)(brow0 + k0 + 64), 0, 1);

    BFrag a0, a1;
    a0.u[0] = *(const uint4*)(arow0 + k0);
    a0.u[1] = *(const uint4*)(arow0 + k0 + 8);
    a1.u[0] = *(const uint4*)(arow1 + k0);
    a1.u[1] = *(const uint4*)(arow1 + k0 + 8);

    const u16* br[4] = { brow0 + k0, brow1 + k0, brow2 + k0, brow3 + k0 };
#pragma unroll
    for (int nt = 0; nt < 4; ++nt) {
      BFrag bfr;
      bfr.u[0] = *(const uint4*)(br[nt]);
      bfr.u[1] = *(const uint4*)(br[nt] + 8);
      acc[0][nt] = __builtin_amdgcn_wmma_f32_16x16x32_bf16(
          false, a0.v, false, bfr.v, (short)0, acc[0][nt], false, false);
      acc[1][nt] = __builtin_amdgcn_wmma_f32_16x16x32_bf16(
          false, a1.v, false, bfr.v, (short)0, acc[1][nt], false, false);
    }
  }
#pragma unroll
  for (int mi = 0; mi < 2; ++mi)
#pragma unroll
    for (int nt = 0; nt < 4; ++nt)
#pragma unroll
      for (int r = 0; r < 8; ++r)
        C[(size_t)(m0 + mi * 16 + r + 8 * hi) * N + n0 + nt * 16 + lo] = acc[mi][nt][r];
}

// ---------------- per-head RMSNorm + RoPE, write transposed bf16 [b][nh][l][hd] ----------------
__global__ __launch_bounds__(32) void qk_post(
    const float* __restrict__ src,    // [b*l][NH*HD] f32 (GEMM output rows)
    const float* __restrict__ lnw,    // [HD]
    u16* __restrict__ dstT, int NH) {
  const int j  = threadIdx.x;                 // 0..31 -> pair (j, j+32)
  const int nh = blockIdx.x % NH;
  const int l  = (blockIdx.x / NH) % L_;
  const int b  =  blockIdx.x / (NH * L_);
  const float* s = src + ((size_t)(b * L_ + l) * NH + nh) * HD_;
  float e1 = s[j], e2 = s[j + 32];
  float ss = e1 * e1 + e2 * e2;
#pragma unroll
  for (int t = 1; t < 32; t <<= 1) ss += __shfl_xor(ss, t, 32);
  const float rms = rsqrtf(ss * (1.0f / 64.0f) + 1e-5f);
  const float n1 = e1 * rms * lnw[j];
  const float n2 = e2 * rms * lnw[j + 32];
  // inv_freq = 1e6^(-j/32) = 2^(-j/32 * log2(1e6))
  const float inv_freq = exp2f(-19.9315685693f * (float)j * (1.0f / 32.0f));
  const float ang = (float)l * inv_freq;
  float sv, cv; sincosf(ang, &sv, &cv);
  u16* d = dstT + ((size_t)(b * NH + nh) * L_ + l) * HD_;
  d[j]      = f2bf(n1 * cv - n2 * sv);
  d[j + 32] = f2bf(n2 * cv + n1 * sv);
}

// ---------------- causal flash attention (GQA) ----------------
// 128 threads = 4 waves; wave w handles 16 query rows [qc*64 + w*16, +16).
__global__ __launch_bounds__(128) void attn_kernel(
    const u16* __restrict__ qT,   // [b][h][l][hd]
    const u16* __restrict__ kT,   // [b][kv][l][hd]
    const u16* __restrict__ vT,   // [b][kv][hd][l]
    u16* __restrict__ attnb) {    // [b][l][H*HD]
  __shared__ __align__(16) u16 Plds[4][16][32];   // per-wave P transpose staging
  const int lane = threadIdx.x & 31;
  const int w    = threadIdx.x >> 5;
  const int lo   = lane & 15, hi = lane >> 4;
  int xb = blockIdx.x;
  const int QC = L_ / 64;
  const int qc = xb % QC; xb /= QC;
  const int h  = xb % H_;
  const int b  = xb / H_;
  const int kv = h / G_;
  const int qbase = qc * 64 + w * 16;

  const u16* qrow = qT + ((size_t)(b * H_  + h )) * L_ * HD_;
  const u16* krow = kT + ((size_t)(b * KV_ + kv)) * L_ * HD_;
  const u16* vrow = vT + ((size_t)(b * KV_ + kv)) * HD_ * L_;

  // Q A-fragments for both 32-wide K-dim steps (held for the whole key loop)
  BFrag aQ[2];
  {
    const u16* qp = qrow + (size_t)(qbase + lo) * HD_;
#pragma unroll
    for (int s = 0; s < 2; ++s) {
      aQ[s].u[0] = *(const uint4*)(qp + 32 * s + 16 * hi);
      aQ[s].u[1] = *(const uint4*)(qp + 32 * s + 16 * hi + 8);
    }
  }

  f32x8 o[4] = {};
  float m[8], lsum[8];
#pragma unroll
  for (int r = 0; r < 8; ++r) { m[r] = -1e30f; lsum[r] = 0.0f; }

  const int kend = qbase + 15;
  for (int kb = 0; kb <= kend; kb += 32) {
    // ---- S = Q * K^T for 32 keys (two 16-key tiles), f32 accum ----
    f32x8 sa = {}, sb = {};
#pragma unroll
    for (int s = 0; s < 2; ++s) {
      BFrag bk0, bk1;
      const u16* kp0 = krow + (size_t)(kb + lo) * HD_ + 32 * s + 16 * hi;
      bk0.u[0] = *(const uint4*)kp0;  bk0.u[1] = *(const uint4*)(kp0 + 8);
      const u16* kp1 = krow + (size_t)(kb + 16 + lo) * HD_ + 32 * s + 16 * hi;
      bk1.u[0] = *(const uint4*)kp1;  bk1.u[1] = *(const uint4*)(kp1 + 8);
      sa = __builtin_amdgcn_wmma_f32_16x16x32_bf16(false, aQ[s].v, false, bk0.v, (short)0, sa, false, false);
      sb = __builtin_amdgcn_wmma_f32_16x16x32_bf16(false, aQ[s].v, false, bk1.v, (short)0, sb, false, false);
    }
    // ---- online softmax (rows live in half-wave lane groups of 16) ----
#pragma unroll
    for (int r = 0; r < 8; ++r) {
      const int qi = qbase + r + 8 * hi;
      float s0 = sa[r] * 0.125f; if (kb + lo      > qi) s0 = -1e30f;
      float s1 = sb[r] * 0.125f; if (kb + 16 + lo > qi) s1 = -1e30f;
      float mx = fmaxf(s0, s1);
#pragma unroll
      for (int t = 1; t < 16; t <<= 1) mx = fmaxf(mx, __shfl_xor(mx, t, 32));
      const float mn = fmaxf(m[r], mx);
      const float al = __expf(m[r] - mn);
      const float p0 = __expf(s0 - mn);
      const float p1 = __expf(s1 - mn);
      float ls = p0 + p1;
#pragma unroll
      for (int t = 1; t < 16; t <<= 1) ls += __shfl_xor(ls, t, 32);
      lsum[r] = lsum[r] * al + ls;
      m[r] = mn;
      o[0][r] *= al; o[1][r] *= al; o[2][r] *= al; o[3][r] *= al;
      Plds[w][r + 8 * hi][lo]      = f2bf(p0);
      Plds[w][r + 8 * hi][16 + lo] = f2bf(p1);
    }
    // wave-private LDS transpose: stores then reads; DS ops are in-order per wave
    asm volatile("s_wait_dscnt 0" ::: "memory");
    BFrag aP;
    {
      const u16* pr = &Plds[w][lo][0];
      aP.u[0] = *(const uint4*)(pr + 16 * hi);
      aP.u[1] = *(const uint4*)(pr + 16 * hi + 8);
    }
    // ---- O += P(16x32) * V(32x64) ----
#pragma unroll
    for (int nt = 0; nt < 4; ++nt) {
      BFrag bv;
      const u16* vp = vrow + (size_t)(nt * 16 + lo) * L_ + kb + 16 * hi;
      bv.u[0] = *(const uint4*)vp;  bv.u[1] = *(const uint4*)(vp + 8);
      o[nt] = __builtin_amdgcn_wmma_f32_16x16x32_bf16(false, aP.v, false, bv.v, (short)0, o[nt], false, false);
    }
  }
  // ---- normalize and store bf16 [b][l][h*64 + dim] ----
#pragma unroll
  for (int r = 0; r < 8; ++r) {
    const float inv = 1.0f / lsum[r];
    u16* d = attnb + ((size_t)(b * L_ + qbase + r + 8 * hi)) * NQ_ + h * HD_;
#pragma unroll
    for (int nt = 0; nt < 4; ++nt)
      d[nt * 16 + lo] = f2bf(o[nt][r] * inv);
  }
}

// ---------------- host launcher ----------------
extern "C" void kernel_launch(void* const* d_in, const int* in_sizes, int n_in,
                              void* d_out, int out_size, void* d_ws, size_t ws_size,
                              hipStream_t stream) {
  (void)in_sizes; (void)n_in; (void)out_size; (void)ws_size;
  const float* x   = (const float*)d_in[0];
  const float* wq  = (const float*)d_in[1];
  const float* wk  = (const float*)d_in[2];
  const float* wv  = (const float*)d_in[3];
  const float* wo  = (const float*)d_in[4];
  const float* qln = (const float*)d_in[5];
  const float* kln = (const float*)d_in[6];

  char* p = (char*)d_ws;
  auto take = [&](size_t bytes) { char* r = p; p += (bytes + 255) & ~(size_t)255; return r; };
  u16*   xb  = (u16*)  take((size_t)M_ * D_  * 2);
  u16*   wqb = (u16*)  take((size_t)NQ_ * D_ * 2);
  u16*   wkb = (u16*)  take((size_t)NK_ * D_ * 2);
  u16*   wvb = (u16*)  take((size_t)NK_ * D_ * 2);
  u16*   wob = (u16*)  take((size_t)D_ * NQ_ * 2);
  float* qf  = (float*)take((size_t)M_ * NQ_ * 4);
  float* kf  = (float*)take((size_t)M_ * NK_ * 4);
  float* vf  = (float*)take((size_t)M_ * NK_ * 4);
  u16*   qT  = (u16*)  take((size_t)B_ * H_  * L_ * HD_ * 2);
  u16*   kT  = (u16*)  take((size_t)B_ * KV_ * L_ * HD_ * 2);
  u16*   vT  = (u16*)  take((size_t)B_ * KV_ * HD_ * L_ * 2);
  u16*   attnb = (u16*)qf;   // alias: qf is dead once qk_post(Q) has run

  int n;
  n = M_  * D_; cast_f32_bf16<<<(n + 255) / 256, 256, 0, stream>>>(x,  xb,  n);
  n = NQ_ * D_; cast_f32_bf16<<<(n + 255) / 256, 256, 0, stream>>>(wq, wqb, n);
  n = NK_ * D_; cast_f32_bf16<<<(n + 255) / 256, 256, 0, stream>>>(wk, wkb, n);
  n = NK_ * D_; cast_f32_bf16<<<(n + 255) / 256, 256, 0, stream>>>(wv, wvb, n);
  n = D_ * NQ_; cast_f32_bf16<<<(n + 255) / 256, 256, 0, stream>>>(wo, wob, n);

  dim3 blk(128);
  gemm_bf16_nt<<<dim3(NQ_ / 64, M_ / 128), blk, 0, stream>>>(xb, wqb, qf, M_, NQ_, D_);
  gemm_bf16_nt<<<dim3(NK_ / 64, M_ / 128), blk, 0, stream>>>(xb, wkb, kf, M_, NK_, D_);
  gemm_bf16_nt<<<dim3(NK_ / 64, M_ / 128), blk, 0, stream>>>(xb, wvb, vf, M_, NK_, D_);

  qk_post<<<B_ * L_ * H_,  32, 0, stream>>>(qf, qln, qT, H_);
  qk_post<<<B_ * L_ * KV_, 32, 0, stream>>>(kf, kln, kT, KV_);
  n = B_ * L_ * KV_ * HD_;
  v_post<<<(n + 255) / 256, 256, 0, stream>>>(vf, vT, n);

  attn_kernel<<<B_ * H_ * (L_ / 64), 128, 0, stream>>>(qT, kT, vT, attnb);

  gemm_bf16_nt<<<dim3(D_ / 64, M_ / 128), blk, 0, stream>>>(attnb, wob, (float*)d_out, M_, D_, NQ_);
}
